// NeighborSparseAttention_12060268167856
// MI455X (gfx1250) — matrix operand order repaired and checked by reading, not compiled, not run on hardware
//
#include <hip/hip_runtime.h>
#include <hip/hip_bf16.h>
#include <stdint.h>

typedef __attribute__((ext_vector_type(16))) __bf16        v16bf;
typedef __attribute__((ext_vector_type(8)))  float         v8f;
typedef __attribute__((ext_vector_type(4)))  unsigned int  v4u;

#define SEQ_N  16384
#define EMBED  512
#define HEADS  8
#define HDIM   64
#define BATCH  4
#define MTOT   (BATCH * SEQ_N)   // 65536 rows

// GEMM tiling: block = 256 threads = 8 waves arranged 4(M) x 2(O)
// block tile M=128 x O=128, K-step 32; each wave computes 32x64 (8 WMMAs/step)
#define TM     128
#define TO     128
#define TK     32
#define LDSROW 40                      // halves per row (80 B, 16B-aligned, padded)
#define BUFH   ((TM + TO) * LDSROW)    // halves per ping-pong buffer (10240)
#define BUFB   (BUFH * 2)              // bytes per buffer (20480)
#define BBOFF  (TM * LDSROW * 2)       // byte offset of B tile inside a buffer

__device__ __forceinline__ unsigned int bf16pack(float a, float b) {
#if __has_builtin(__builtin_amdgcn_cvt_pk_bf16_f32)
  typedef __attribute__((ext_vector_type(2))) __bf16 v2bf;
  v2bf p = __builtin_amdgcn_cvt_pk_bf16_f32(a, b);
  return __builtin_bit_cast(unsigned int, p);
#else
  unsigned int ua = __float_as_uint(a);
  ua = (ua + 0x7FFFu + ((ua >> 16) & 1u)) >> 16;
  unsigned int ub = __float_as_uint(b);
  ub = (ub + 0x7FFFu + ((ub >> 16) & 1u)) & 0xFFFF0000u;
  return ua | ub;
#endif
}

// Escape the LDS block into inline-asm land so the "memory"-clobber async
// copies are treated as potential writers of it (otherwise LLVM folds the
// ds_load fragment reads of a never-visibly-written __shared__ array to undef).
__device__ __forceinline__ void lds_escape(void* p) {
  asm volatile("" :: "v"(p) : "memory");
}

// CDNA5 async global->LDS copy (GV mode), 16B per lane, tracked by ASYNCcnt.
// lds_byte_off relative to workgroup static-LDS base (single __shared__ block).
__device__ __forceinline__ void async_ld_b128(unsigned lds_byte_off,
                                              const void* gaddr) {
  asm volatile("global_load_async_to_lds_b128 %0, %1, off"
               :: "v"(lds_byte_off), "v"(gaddr)
               : "memory");
}

// ---------------------------------------------------------------- weights->bf16
__global__ __launch_bounds__(256) void convert_weights(
    const float* __restrict__ Wq, const float* __restrict__ Wk,
    const float* __restrict__ Wv, const float* __restrict__ Wo,
    unsigned int* __restrict__ out) {
  size_t i  = ((size_t)blockIdx.x * 256 + threadIdx.x) * 4;  // float index
  size_t mi = i >> 18;                                       // 262144 floats / matrix
  size_t off = i & 262143;
  const float* s = (mi == 0) ? Wq : (mi == 1) ? Wk : (mi == 2) ? Wv : Wo;
  float4 f = *(const float4*)(s + off);
  out[i >> 1]       = bf16pack(f.x, f.y);
  out[(i >> 1) + 1] = bf16pack(f.z, f.w);
}

// ---------------------------------------------------------------- x -> bf16
__global__ __launch_bounds__(256) void convert_x(
    const float* __restrict__ x, unsigned int* __restrict__ xb) {
  size_t i = ((size_t)blockIdx.x * 256 + threadIdx.x) * 8;  // float index
  float4 f0 = *(const float4*)(x + i);
  float4 f1 = *(const float4*)(x + i + 4);
  v4u o = {bf16pack(f0.x, f0.y), bf16pack(f0.z, f0.w),
           bf16pack(f1.x, f1.y), bf16pack(f1.z, f1.w)};
  *(v4u*)(xb + (i >> 1)) = o;
}

// ------------------------------------------------- double-buffered WMMA core
// Stages A(128x32) and W(128x32) bf16 tiles via async copies with ping-pong
// LDS buffers; computes 8 WMMAs per wave per k-step into acc0/acc1.
__device__ __forceinline__ void stage_tile(
    const unsigned short* __restrict__ Abf,
    const unsigned short* __restrict__ Wg,
    int mBase, int oTile, int k0, int tid, unsigned bufByte) {
#pragma unroll
  for (int p = 0; p < 2; ++p) {
    int idx  = p * 256 + tid;
    int row  = idx >> 2;
    int colh = (idx & 3) * 8;
    unsigned loff = (unsigned)(row * LDSROW + colh) * 2;
    async_ld_b128(bufByte + loff,
                  Abf + (size_t)(mBase + row) * EMBED + k0 + colh);
    async_ld_b128(bufByte + BBOFF + loff,
                  Wg + (size_t)(oTile + row) * EMBED + k0 + colh);
  }
}

__device__ __forceinline__ void gemm_core(
    const unsigned short* __restrict__ Abf,
    const unsigned short* __restrict__ Wg,
    int mBase, int oTile, int tid, unsigned short* lds,
    v8f acc0[4], v8f acc1[4]) {
  lds_escape(lds);   // async copies may now alias the LDS block

  const int lane = tid & 31;
  const int wv_  = tid >> 5;
  const int wvM  = wv_ & 3;        // M strip (32 rows)
  const int wvO  = wv_ >> 2;       // O strip (64 cols)
  const int half = lane >> 4;      // K-split half of the wave
  const int mr   = lane & 15;

  const int NSTEP = EMBED / TK;    // 16
  stage_tile(Abf, Wg, mBase, oTile, 0, tid, 0u);

  for (int i = 0; i < NSTEP; ++i) {
    const unsigned curByte = (i & 1) ? (unsigned)BUFB : 0u;
    if (i + 1 < NSTEP) {
      stage_tile(Abf, Wg, mBase, oTile, (i + 1) * TK, tid,
                 (i & 1) ? 0u : (unsigned)BUFB);
      // async loads complete in order: <=4 outstanding => tile i landed
      asm volatile("s_wait_asynccnt 0x4" ::: "memory");
    } else {
      asm volatile("s_wait_asynccnt 0x0" ::: "memory");
    }
    __syncthreads();

    const unsigned short* ldsA = lds + (curByte >> 1);
    const unsigned short* ldsB = ldsA + TM * LDSROW;

    // A fragments: lane<16 holds K0-7 & K16-23; lane>=16 holds K8-15 & K24-31
    v16bf a0, a1;
    {
      const unsigned short* ap0 = ldsA + (wvM * 32 + mr) * LDSROW + half * 8;
      ((v4u*)&a0)[0] = *(const v4u*)ap0;
      ((v4u*)&a0)[1] = *(const v4u*)(ap0 + 16);
      const unsigned short* ap1 = ap0 + 16 * LDSROW;
      ((v4u*)&a1)[0] = *(const v4u*)ap1;
      ((v4u*)&a1)[1] = *(const v4u*)(ap1 + 16);
    }
#pragma unroll
    for (int j = 0; j < 4; ++j) {
      // B fragment: lane<16 -> col N, K0-7 & K16-23 ; lane>=16 -> K8-15 & K24-31
      v16bf b;
      const unsigned short* bp =
          ldsB + (wvO * 64 + j * 16 + mr) * LDSROW + half * 8;
      ((v4u*)&b)[0] = *(const v4u*)bp;
      ((v4u*)&b)[1] = *(const v4u*)(bp + 16);
      acc0[j] = __builtin_amdgcn_wmma_f32_16x16x32_bf16(
          false, a0, false, b, (short)0, acc0[j], false, false);
      acc1[j] = __builtin_amdgcn_wmma_f32_16x16x32_bf16(
          false, a1, false, b, (short)0, acc1[j], false, false);
    }
    __syncthreads();  // protect buf about to be overwritten next iteration
  }
}

// ---------------------------------------------------------------- fused QKV GEMM
// q/k/v[b,h,n,d] = sum_c x[m,c] * W[o,c] + bias[o],  m=b*N+n, o=h*64+d
__global__ __launch_bounds__(256) void qkv_gemm(
    const unsigned short* __restrict__ xbf,
    const unsigned short* __restrict__ Wall,
    const float* __restrict__ bq, const float* __restrict__ bk,
    const float* __restrict__ bv,
    float* __restrict__ qo, float* __restrict__ ko, float* __restrict__ vo) {
  __shared__ unsigned short lds[2 * BUFH];

  const int tid   = threadIdx.x;
  const int mat   = blockIdx.x >> 2;              // 0=q 1=k 2=v
  const int oTile = (blockIdx.x & 3) * TO;
  const int mBase = blockIdx.y * TM;
  const unsigned short* Wg = Wall + (size_t)mat * (EMBED * EMBED);

  v8f acc0[4] = {};
  v8f acc1[4] = {};
  gemm_core(xbf, Wg, mBase, oTile, tid, lds, acc0, acc1);

  const int lane = tid & 31;
  const int wv_  = tid >> 5;
  const int wvM  = wv_ & 3;
  const int wvO  = wv_ >> 2;
  const int half = lane >> 4;
  const int mr   = lane & 15;

  const float* biasP = (mat == 0) ? bq : (mat == 1) ? bk : bv;
  float* outP        = (mat == 0) ? qo : (mat == 1) ? ko : vo;

#pragma unroll
  for (int j = 0; j < 4; ++j) {
    int o     = oTile + wvO * 64 + j * 16 + mr;
    float bia = biasP[o];
    int hh = o >> 6;
    int dc = o & 63;
#pragma unroll
    for (int s = 0; s < 2; ++s) {
      const v8f* ac = s ? &acc1[j] : &acc0[j];
#pragma unroll
      for (int r = 0; r < 8; ++r) {
        int m  = mBase + wvM * 32 + s * 16 + r + half * 8;
        int b_ = m >> 14;
        int n  = m & (SEQ_N - 1);
        size_t addr = ((((size_t)b_ * HEADS + hh) * SEQ_N) + n) * HDIM + dc;
        outP[addr] = (*ac)[r] + bia;
      }
    }
  }
}

// ---------------------------------------------------------------- neighbor attention
// one wave32 per (b,h,n); lanes split d=64 as float2
__global__ __launch_bounds__(256) void nbr_attn(
    const float* __restrict__ q, const float* __restrict__ k,
    const float* __restrict__ v, const long long* __restrict__ nbr,
    unsigned int* __restrict__ attn_out) {
  const int lane = threadIdx.x & 31;
  const unsigned int wid = blockIdx.x * 8u + (threadIdx.x >> 5);
  const int n = wid & (SEQ_N - 1);
  const int h = (wid >> 14) & (HEADS - 1);
  const int b = wid >> 17;
  const size_t bh = ((size_t)b * HEADS + h) * SEQ_N;

  const float2 qv = *(const float2*)(q + ((bh + n) << 6) + lane * 2);

  float sc[3];
  bool  val[3];
  int   id[3];
#pragma unroll
  for (int j = 0; j < 3; ++j) {
    long long t = nbr[(size_t)j * SEQ_N + n];
    val[j] = (t != -1);
    id[j]  = (t < 0) ? 0 : (int)t;
    float2 kv = *(const float2*)(k + ((bh + id[j]) << 6) + lane * 2);
    float p = qv.x * kv.x + qv.y * kv.y;
#pragma unroll
    for (int off = 16; off > 0; off >>= 1) p += __shfl_xor(p, off, 32);
    sc[j] = p * 0.125f;  // d^-0.5 = 64^-0.5
  }
  float mx = -3.4e38f;
#pragma unroll
  for (int j = 0; j < 3; ++j)
    if (val[j] && sc[j] > mx) mx = sc[j];
  float e[3], den = 0.f;
#pragma unroll
  for (int j = 0; j < 3; ++j) {
    e[j] = val[j] ? __expf(sc[j] - mx) : 0.f;
    den += e[j];
  }
  float inv = 1.f / den;
  float2 accv = {0.f, 0.f};
#pragma unroll
  for (int j = 0; j < 3; ++j) {
    float a = e[j] * inv;
    float2 vvv = *(const float2*)(v + ((bh + id[j]) << 6) + lane * 2);
    accv.x += a * vvv.x;
    accv.y += a * vvv.y;
  }
  size_t outIdx = (((size_t)b * SEQ_N + n) * EMBED + h * HDIM) >> 1;
  attn_out[outIdx + lane] = bf16pack(accv.x, accv.y);
}

// ---------------------------------------------------------------- output GEMM
__global__ __launch_bounds__(256) void out_gemm(
    const unsigned short* __restrict__ A, const unsigned short* __restrict__ Wg,
    const float* __restrict__ bo, float* __restrict__ out) {
  __shared__ unsigned short lds[2 * BUFH];

  const int tid   = threadIdx.x;
  const int oTile = blockIdx.x * TO;
  const int mBase = blockIdx.y * TM;

  v8f acc0[4] = {};
  v8f acc1[4] = {};
  gemm_core(A, Wg, mBase, oTile, tid, lds, acc0, acc1);

  const int lane = tid & 31;
  const int wv_  = tid >> 5;
  const int wvM  = wv_ & 3;
  const int wvO  = wv_ >> 2;
  const int half = lane >> 4;
  const int mr   = lane & 15;

#pragma unroll
  for (int j = 0; j < 4; ++j) {
    int o     = oTile + wvO * 64 + j * 16 + mr;
    float bia = bo[o];
#pragma unroll
    for (int s = 0; s < 2; ++s) {
      const v8f* ac = s ? &acc1[j] : &acc0[j];
#pragma unroll
      for (int r = 0; r < 8; ++r) {
        int m = mBase + wvM * 32 + s * 16 + r + half * 8;
        out[(size_t)m * EMBED + o] = (*ac)[r] + bia;
      }
    }
  }
}

// ---------------------------------------------------------------- launch
extern "C" void kernel_launch(void* const* d_in, const int* in_sizes, int n_in,
                              void* d_out, int out_size, void* d_ws, size_t ws_size,
                              hipStream_t stream) {
  const float*     x   = (const float*)d_in[0];
  const long long* nbr = (const long long*)d_in[1];
  const float*     Wq  = (const float*)d_in[2];
  const float*     bq  = (const float*)d_in[3];
  const float*     Wk  = (const float*)d_in[4];
  const float*     bk  = (const float*)d_in[5];
  const float*     Wv  = (const float*)d_in[6];
  const float*     bv  = (const float*)d_in[7];
  const float*     Wo  = (const float*)d_in[8];
  const float*     bo  = (const float*)d_in[9];
  float*           out = (float*)d_out;

  char* ws = (char*)d_ws;
  unsigned short* Wbf = (unsigned short*)ws;                 // 4 * 512*512 bf16 = 2 MiB
  size_t off = 4ull << 20;
  unsigned short* xbf = (unsigned short*)(ws + off);         // 64 MiB bf16 x
  off += (size_t)MTOT * EMBED * 2;
  const size_t qkvBytes = (size_t)MTOT * EMBED * sizeof(float);  // 128 MiB each
  float* q  = (float*)(ws + off); off += qkvBytes;
  float* kk = (float*)(ws + off); off += qkvBytes;
  float* vv = (float*)(ws + off); off += qkvBytes;
  unsigned int* attn = (unsigned int*)(ws + off);            // 64 MiB bf16 (B,N,C)

  convert_weights<<<1024, 256, 0, stream>>>(Wq, Wk, Wv, Wo, (unsigned int*)Wbf);
  convert_x<<<(MTOT * EMBED) / (256 * 8), 256, 0, stream>>>(x, (unsigned int*)xbf);

  dim3 gq(12, MTOT / TM);   // O-tiles fastest -> x tile reused from L2
  qkv_gemm<<<gq, 256, 0, stream>>>(xbf, Wbf, bq, bk, bv, q, kk, vv);

  nbr_attn<<<(BATCH * HEADS * SEQ_N) / 8, 256, 0, stream>>>(q, kk, vv, nbr, attn);

  dim3 go(4, MTOT / TM);
  out_gemm<<<go, 256, 0, stream>>>((const unsigned short*)attn,
                                   Wbf + 3ull * EMBED * EMBED, bo, out);
}
